// MultiHeadAttention_15960098472130
// MI455X (gfx1250) — compile-verified, matrix operand order checked by reading, not run on hardware
//
#include <hip/hip_runtime.h>
#include <hip/hip_bf16.h>

// ---------- CDNA5 WMMA types ----------
typedef __attribute__((ext_vector_type(16))) __bf16 v16bf;
typedef __attribute__((ext_vector_type(8)))  __bf16 v8bf;
typedef __attribute__((ext_vector_type(8)))  float  v8f;

// D = A(16x32 bf16) x B(32x16 bf16) + C(16x16 f32)
__device__ __forceinline__ v8f wmma_bf16(v16bf a, v16bf b, v8f c) {
    return __builtin_amdgcn_wmma_f32_16x16x32_bf16(false, a, false, b,
                                                   (short)0, c, false, false);
}

// 16x32 bf16 fragment (A-operand layout, ISA 7.12.2):
// lane L in [0,15]: row=L, K={0..7,16..23}; lane L+16: row=L, K={8..15,24..31}.
__device__ __forceinline__ v16bf load_frag(const __bf16* base, int ld) {
    const int lane = threadIdx.x & 31;
    const int row  = lane & 15;
    const int kh   = (lane >> 4) & 1;
    const __bf16* p = base + row * ld + kh * 8;
    v8bf lo = *reinterpret_cast<const v8bf*>(p);        // 16B
    v8bf hi = *reinterpret_cast<const v8bf*>(p + 16);   // 16B
    return __builtin_shufflevector(lo, hi, 0,1,2,3,4,5,6,7,8,9,10,11,12,13,14,15);
}

// gfx1250 async global->LDS copy (ASYNCcnt path). Low 32 bits of a generic
// pointer into __shared__ are the LDS byte address (aperture rule).
__device__ __forceinline__ void async_ld16(void* lds_dst, const void* gsrc) {
    const unsigned loff = (unsigned)(uintptr_t)lds_dst;
    asm volatile("global_load_async_to_lds_b128 %0, %1, off"
                 :: "v"(loff), "v"((unsigned long long)(uintptr_t)gsrc)
                 : "memory");
}
__device__ __forceinline__ void wait_async0() {
    asm volatile("s_wait_asynccnt 0x0" ::: "memory");
}

// Reductions across a 16-lane half of the wave.
__device__ __forceinline__ float half_max(float v) {
#pragma unroll
    for (int m = 1; m < 16; m <<= 1) v = fmaxf(v, __shfl_xor(v, m, 32));
    return v;
}
__device__ __forceinline__ float half_sum(float v) {
#pragma unroll
    for (int m = 1; m < 16; m <<= 1) v += __shfl_xor(v, m, 32);
    return v;
}

// ---------------- problem dims ----------------
#define DMODEL 1024
#define SEQ    2048
#define NHEAD  16
#define HDIM   64

// =================================================================
// Kernel 1: fused QKV projection with LDS double buffering.
// q/k/v[b,h,s,hd] (bf16) = x @ W^T.  256 thr (8 waves); tile M=64,N=64 x3 mats.
// =================================================================
__global__ __launch_bounds__(256) void qkv_kernel(
    const float* __restrict__ X, const float* __restrict__ Wq,
    const float* __restrict__ Wk, const float* __restrict__ Wv,
    __bf16* __restrict__ Qo, __bf16* __restrict__ Ko, __bf16* __restrict__ Vo)
{
    __shared__ __attribute__((aligned(16))) __bf16 sm[2][4 * 64 * 32]; // ping-pong: Xs + 3xWs

    const int tid  = threadIdx.x;
    const int lane = tid & 31;
    const int wave = tid >> 5;
    const int n0 = blockIdx.x * 64;
    const int m0 = blockIdx.y * 64;
    const int mt = wave >> 1;      // 16-row slice
    const int np = wave & 1;       // 32-col slice

    const float* Wm[3] = {Wq, Wk, Wv};
    __bf16*      Om[3] = {Qo, Ko, Vo};

    // register staging for the next tile (hides global latency under WMMA)
    float4 rx[2];
    float4 rw[3][2];

    auto stage_load = [&](int k0) {
#pragma unroll
        for (int u = 0; u < 2; ++u) {
            const int i = tid + u * 256;
            const int row = i >> 3, c4 = (i & 7) << 2;
            rx[u] = *reinterpret_cast<const float4*>(X + (size_t)(m0 + row) * DMODEL + k0 + c4);
#pragma unroll
            for (int mat = 0; mat < 3; ++mat)
                rw[mat][u] = *reinterpret_cast<const float4*>(Wm[mat] + (size_t)(n0 + row) * DMODEL + k0 + c4);
        }
    };
    auto stage_store = [&](int b) {
        __bf16* Xs = &sm[b][0];
        __bf16* Ws = &sm[b][2048];
#pragma unroll
        for (int u = 0; u < 2; ++u) {
            const int i = tid + u * 256;
            const int row = i >> 3, c4 = (i & 7) << 2;
            __bf16* d = Xs + row * 32 + c4;
            d[0] = (__bf16)rx[u].x; d[1] = (__bf16)rx[u].y;
            d[2] = (__bf16)rx[u].z; d[3] = (__bf16)rx[u].w;
#pragma unroll
            for (int mat = 0; mat < 3; ++mat) {
                __bf16* dw = Ws + mat * 2048 + row * 32 + c4;
                dw[0] = (__bf16)rw[mat][u].x; dw[1] = (__bf16)rw[mat][u].y;
                dw[2] = (__bf16)rw[mat][u].z; dw[3] = (__bf16)rw[mat][u].w;
            }
        }
    };

    v8f acc[3][2];
#pragma unroll
    for (int a = 0; a < 3; ++a)
#pragma unroll
        for (int t = 0; t < 2; ++t)
#pragma unroll
            for (int i = 0; i < 8; ++i) acc[a][t][i] = 0.0f;

    stage_load(0);
    stage_store(0);
    __syncthreads();

#pragma unroll 1
    for (int kc = 0; kc < DMODEL / 32; ++kc) {
        if (kc + 1 < DMODEL / 32) stage_load((kc + 1) * 32);   // global loads in flight

        const __bf16* Xs = &sm[kc & 1][0];
        const __bf16* Ws = &sm[kc & 1][2048];
        // load all fragments into distinct registers, then issue the WMMA chain
        const v16bf af = load_frag(Xs + mt * 16 * 32, 32);
        v16bf bfr[3][2];
#pragma unroll
        for (int mat = 0; mat < 3; ++mat)
#pragma unroll
            for (int t = 0; t < 2; ++t)
                bfr[mat][t] = load_frag(Ws + mat * 2048 + (np * 32 + t * 16) * 32, 32);
#pragma unroll
        for (int mat = 0; mat < 3; ++mat)
#pragma unroll
            for (int t = 0; t < 2; ++t)
                acc[mat][t] = wmma_bf16(af, bfr[mat][t], acc[mat][t]);

        __syncthreads();                       // everyone done reading buf[kc&1]
        if (kc + 1 < DMODEL / 32) {
            stage_store((kc + 1) & 1);
            __syncthreads();                   // stores visible before next compute
        }
    }

    // scatter into [B,H,S,HD]
    const int half = lane >> 4, n15 = lane & 15;
#pragma unroll
    for (int mat = 0; mat < 3; ++mat)
#pragma unroll
        for (int t = 0; t < 2; ++t)
#pragma unroll
            for (int j = 0; j < 8; ++j) {
                const int m = m0 + mt * 16 + j + 8 * half;   // 0..4095
                const int n = n0 + np * 32 + t * 16 + n15;   // 0..1023
                const int b = m >> 11, s = m & (SEQ - 1);
                const int h = n >> 6, hd = n & (HDIM - 1);
                Om[mat][(((size_t)b * NHEAD + h) * SEQ + s) * HDIM + hd] = (__bf16)acc[mat][t][j];
            }
}

// =================================================================
// Kernel 2: causal flash attention per (b,h). 128 thr = 4 waves, each wave
// a 16-row Q tile; 32-key blocks streamed through LDS (K via async-to-LDS),
// online softmax, all matmuls via WMMA.
// =================================================================
__global__ __launch_bounds__(128) void attn_kernel(
    const __bf16* __restrict__ Q, const __bf16* __restrict__ K,
    const __bf16* __restrict__ V, __bf16* __restrict__ O)
{
    __shared__ __attribute__((aligned(16))) __bf16 Ks[32 * 64];      // K rows
    __shared__ __attribute__((aligned(16))) __bf16 Vt[64 * 32];      // V^T: [d][k]
    __shared__ __attribute__((aligned(16))) __bf16 Pb[4 * 16 * 32];  // per-wave P staging

    const int tid  = threadIdx.x;
    const int lane = tid & 31;
    const int wave = tid >> 5;
    const int half = lane >> 4, n15 = lane & 15;

    const int bh = blockIdx.y;               // b*H + h
    const int b  = bh >> 4, h = bh & 15;
    const int q0 = blockIdx.x * 64;
    const int qbase = q0 + wave * 16;

    const __bf16* qptr = Q + ((size_t)bh * SEQ + qbase) * HDIM;
    const v16bf qf0 = load_frag(qptr, HDIM);        // K-dim 0..31
    const v16bf qf1 = load_frag(qptr + 32, HDIM);   // K-dim 32..63

    v8f o[4];
#pragma unroll
    for (int t = 0; t < 4; ++t)
#pragma unroll
        for (int i = 0; i < 8; ++i) o[t][i] = 0.0f;
    float mrow[8], lrow[8];
#pragma unroll
    for (int j = 0; j < 8; ++j) { mrow[j] = -3.0e38f; lrow[j] = 0.0f; }

    const __bf16* kbase = K + (size_t)bh * SEQ * HDIM;
    const __bf16* vbase = V + (size_t)bh * SEQ * HDIM;
    const float scale = 0.125f;              // 1/sqrt(64)
    const int nkb = q0 / 32 + 2;             // causal

    for (int kb = 0; kb < nkb; ++kb) {
        const int k0 = kb * 32;
        __syncthreads();
        // stage 32 K rows (contiguous 4KB, already bf16): async global->LDS
        {
            const __bf16* src = kbase + (size_t)k0 * HDIM;
#pragma unroll
            for (int u = 0; u < 2; ++u) {
                const int i = tid + u * 128;          // 256 x 16B vectors
                async_ld16(Ks + i * 8, src + i * 8);
            }
        }
        // stage V transposed: Vt[d][r] = V[k0+r][d]  (needs VGPR transpose)
#pragma unroll
        for (int ii = 0; ii < 16; ++ii) {
            const int idx = ii * 128 + tid;          // 0..2047
            const int r = idx >> 6, d = idx & 63;
            Vt[d * 32 + r] = vbase[(size_t)(k0 + r) * HDIM + d];
        }
        wait_async0();
        __syncthreads();

        // scores: S(16x32) = Q(16x64) @ K^T  (all frags loaded up front)
        const v16bf kf0 = load_frag(Ks, 64);
        const v16bf kf1 = load_frag(Ks + 32, 64);
        const v16bf kf2 = load_frag(Ks + 16 * 64, 64);
        const v16bf kf3 = load_frag(Ks + 16 * 64 + 32, 64);
        v8f s0, s1;
#pragma unroll
        for (int i = 0; i < 8; ++i) { s0[i] = 0.0f; s1[i] = 0.0f; }
        s0 = wmma_bf16(qf0, kf0, s0);
        s0 = wmma_bf16(qf1, kf1, s0);
        s1 = wmma_bf16(qf0, kf2, s1);
        s1 = wmma_bf16(qf1, kf3, s1);

        // online softmax (rows live in 16-lane halves)
        float p0[8], p1[8], alpha[8];
#pragma unroll
        for (int j = 0; j < 8; ++j) {
            const int qg = qbase + j + 8 * half;
            float a  = s0[j] * scale;
            float bs = s1[j] * scale;
            if (k0 + n15 > qg)      a  = -1.0e30f;   // causal mask
            if (k0 + 16 + n15 > qg) bs = -1.0e30f;
            const float rmax = half_max(fmaxf(a, bs));
            const float mnew = fmaxf(mrow[j], rmax);
            const float e0 = __expf(a - mnew);
            const float e1 = __expf(bs - mnew);
            const float rsum = half_sum(e0 + e1);
            alpha[j] = __expf(mrow[j] - mnew);
            lrow[j]  = lrow[j] * alpha[j] + rsum;
            mrow[j]  = mnew;
            p0[j] = e0; p1[j] = e1;
        }

        // C-layout -> A-fragment layout via per-wave LDS staging
        __bf16* Pw = Pb + wave * (16 * 32);
#pragma unroll
        for (int j = 0; j < 8; ++j) {
            const int m = j + 8 * half;
            Pw[m * 32 + n15]      = (__bf16)p0[j];
            Pw[m * 32 + 16 + n15] = (__bf16)p1[j];
        }
        __syncthreads();

        // rescale running O, then O += P(16x32) @ V(32x64)
#pragma unroll
        for (int t = 0; t < 4; ++t)
#pragma unroll
            for (int j = 0; j < 8; ++j) o[t][j] *= alpha[j];
        const v16bf pf  = load_frag(Pw, 32);
        const v16bf vf0 = load_frag(Vt + 0 * 16 * 32, 32);
        const v16bf vf1 = load_frag(Vt + 1 * 16 * 32, 32);
        const v16bf vf2 = load_frag(Vt + 2 * 16 * 32, 32);
        const v16bf vf3 = load_frag(Vt + 3 * 16 * 32, 32);
        o[0] = wmma_bf16(pf, vf0, o[0]);
        o[1] = wmma_bf16(pf, vf1, o[1]);
        o[2] = wmma_bf16(pf, vf2, o[2]);
        o[3] = wmma_bf16(pf, vf3, o[3]);
    }

    // normalize, write [B,S,D] bf16 for the output projection
#pragma unroll
    for (int j = 0; j < 8; ++j) lrow[j] = 1.0f / lrow[j];
#pragma unroll
    for (int t = 0; t < 4; ++t)
#pragma unroll
        for (int j = 0; j < 8; ++j) {
            const int qg = qbase + j + 8 * half;
            const int d  = t * 16 + n15;
            O[((size_t)b * SEQ + qg) * DMODEL + h * HDIM + d] = (__bf16)(o[t][j] * lrow[j]);
        }
}

// =================================================================
// Kernel 3: out = attn_out @ Wo^T + bo (f32 out). A tile staged via
// async global->LDS (bf16 source); Wo converted f32->bf16 through VGPRs.
// =================================================================
__global__ __launch_bounds__(256) void oproj_kernel(
    const __bf16* __restrict__ A, const float* __restrict__ Wo,
    const float* __restrict__ bo, float* __restrict__ C)
{
    __shared__ __attribute__((aligned(16))) __bf16 As[64 * 32];
    __shared__ __attribute__((aligned(16))) __bf16 Ws[64 * 32];

    const int tid  = threadIdx.x;
    const int lane = tid & 31;
    const int wave = tid >> 5;
    const int n0 = blockIdx.x * 64, m0 = blockIdx.y * 64;
    const int mt = wave >> 1, np = wave & 1;

    v8f acc[2];
#pragma unroll
    for (int t = 0; t < 2; ++t)
#pragma unroll
        for (int i = 0; i < 8; ++i) acc[t][i] = 0.0f;

#pragma unroll 1
    for (int kc = 0; kc < DMODEL / 32; ++kc) {
        const int k0 = kc * 32;
        __syncthreads();
        { // A tile: 256 x 16B async copies, one per thread
            const int row = tid >> 2, c8 = (tid & 3) << 3;
            async_ld16(As + row * 32 + c8,
                       A + (size_t)(m0 + row) * DMODEL + k0 + c8);
        }
        // Wo tile f32->bf16, two straight-line iterations
#pragma unroll
        for (int u = 0; u < 2; ++u) {
            const int i = tid + u * 256;
            const int row = i >> 3, c4 = (i & 7) << 2;
            float4 f = *reinterpret_cast<const float4*>(Wo + (size_t)(n0 + row) * DMODEL + k0 + c4);
            __bf16* d = Ws + row * 32 + c4;
            d[0] = (__bf16)f.x; d[1] = (__bf16)f.y; d[2] = (__bf16)f.z; d[3] = (__bf16)f.w;
        }
        wait_async0();
        __syncthreads();

        const v16bf af  = load_frag(As + mt * 16 * 32, 32);
        const v16bf bf0 = load_frag(Ws + (np * 32) * 32, 32);
        const v16bf bf1 = load_frag(Ws + (np * 32 + 16) * 32, 32);
        acc[0] = wmma_bf16(af, bf0, acc[0]);
        acc[1] = wmma_bf16(af, bf1, acc[1]);
    }

    const int half = lane >> 4, n15 = lane & 15;
#pragma unroll
    for (int t = 0; t < 2; ++t)
#pragma unroll
        for (int j = 0; j < 8; ++j) {
            const int m = m0 + mt * 16 + j + 8 * half;
            const int n = n0 + np * 32 + t * 16 + n15;
            C[(size_t)m * DMODEL + n] = acc[t][j] + bo[n];
        }
}

// =================================================================
extern "C" void kernel_launch(void* const* d_in, const int* in_sizes, int n_in,
                              void* d_out, int out_size, void* d_ws, size_t ws_size,
                              hipStream_t stream) {
    const float* x  = (const float*)d_in[0];
    const float* Wq = (const float*)d_in[1];
    const float* Wk = (const float*)d_in[2];
    const float* Wv = (const float*)d_in[3];
    const float* Wo = (const float*)d_in[4];
    const float* bo = (const float*)d_in[5];
    float* out = (float*)d_out;

    // workspace carve: Q,K,V,O as bf16 (4 x 8 MB = 32 MB)
    const size_t NQ = (size_t)2 * NHEAD * SEQ * HDIM;   // 4,194,304 elems
    __bf16* q = (__bf16*)d_ws;
    __bf16* k = q + NQ;
    __bf16* v = k + NQ;
    __bf16* o = v + NQ;

    qkv_kernel<<<dim3(DMODEL / 64, 4096 / 64), 256, 0, stream>>>(x, Wq, Wk, Wv, q, k, v);
    attn_kernel<<<dim3(SEQ / 64, 2 * NHEAD), 128, 0, stream>>>(q, k, v, o);
    oproj_kernel<<<dim3(DMODEL / 64, 4096 / 64), 256, 0, stream>>>(o, Wo, bo, out);
}
